// GPTModel_36086315221269
// MI455X (gfx1250) — matrix operand order repaired and checked by reading, not compiled, not run on hardware
//
#include <hip/hip_runtime.h>
#include <cstdint>
#include <cstddef>

// GPT-2 small forward + loss on gfx1250 (CDNA5, wave32, WMMA bf16).
#define Bn   4
#define Tn   1024
#define Vn   50257
#define Vpad 50304          // padded vocab: multiple of 64
#define Cn   768
#define Hn   12
#define Dn   64
#define Ln   12
#define FFn  3072
#define Mn   (Bn * Tn)      // 4096 token rows
#define EPSf 1e-5f

typedef __attribute__((ext_vector_type(16))) __bf16     v16bf;
typedef __attribute__((ext_vector_type(8)))  float      v8f;
typedef __attribute__((ext_vector_type(4)))  unsigned   u32x4;
typedef unsigned short ushort_t;

static __device__ __forceinline__ ushort_t f2bf(float f) {
  unsigned u = __float_as_uint(f);
  u += 0x7fffu + ((u >> 16) & 1u);          // round-to-nearest-even
  return (ushort_t)(u >> 16);
}

// Build a 16-half fragment from two contiguous 8-half (16B) runs.
static __device__ __forceinline__ v16bf ld_frag(const ushort_t* p0, const ushort_t* p1) {
  union { u32x4 u[2]; v16bf v; } r;
  r.u[0] = *(const u32x4*)p0;
  r.u[1] = *(const u32x4*)p1;
  return r.v;
}

static __device__ __forceinline__ v8f wmma_bf16(v16bf a, v16bf b, v8f c) {
  return __builtin_amdgcn_wmma_f32_16x16x32_bf16(false, a, false, b, (short)0, c,
                                                 false, false);
}

// ---------------------------------------------------------------- embedding
__global__ __launch_bounds__(256) void embed_kernel(const int* __restrict__ x,
                                                    const float* __restrict__ tok,
                                                    const float* __restrict__ pos,
                                                    float* __restrict__ h,
                                                    ushort_t* __restrict__ xb) {
  int idx = blockIdx.x * 256 + threadIdx.x;          // exact: Mn*Cn / 256 blocks
  int m = idx / Cn, c = idx % Cn;
  int t = x[m];
  float v = tok[(size_t)t * Cn + c] + pos[(size_t)t * Cn + c];  // quirk: pos[x]
  h[idx]  = v;
  xb[idx] = f2bf(v);
}

// ------------------------------------------------- weight pack (fp32 -> bf16)
// Packed B layout: tiles of 32(K)x16(N). Tile (kt,nt) at ((nt*KT+kt)*512) halves.
// Within a tile: lane l holds K = kt*32 + (l>>4)*16 + j (j=0..15), N = nt*16 + (l&15),
// stored contiguously at l*16 + j  (contiguous k order, same as the GEMM's A fragments).
template <int HEADLAYOUT>
__global__ __launch_bounds__(256) void pack_kernel(const float* __restrict__ src,
                                                   ushort_t* __restrict__ dst,
                                                   int K, int Nsrc) {
  int e    = blockIdx.x * 256 + threadIdx.x;         // e in [0, K*Npad)
  int tile = e >> 9;
  int rem  = e & 511;
  int lane = rem >> 4, j = rem & 15;
  int KT   = K >> 5;
  int nt   = tile / KT, kt = tile % KT;
  int k    = kt * 32 + (lane >> 4) * 16 + j;
  int n    = nt * 16 + (lane & 15);
  float v  = 0.f;
  if (HEADLAYOUT) {                                  // src shape [H, C, D] -> (k=c, n=h*64+d)
    v = src[(size_t)(n >> 6) * (Cn * Dn) + (size_t)k * Dn + (n & 63)];
  } else if (n < Nsrc) {                             // row-major [K, Nsrc]
    v = src[(size_t)k * Nsrc + n];
  }
  dst[e] = f2bf(v);
}

// ------------------------------------------------------------ generic GEMM
// A: bf16 row-major [M x K]; Bp: packed bf16 [K x N]; wave -> 32x64 output
// (two 16-row blocks sharing the same B fragments: 8 WMMAs per 32-K step).
// Block = 128 threads (4 waves stacked in M); grid = (N/64, M/128).
enum { E_QKV_TD = 0, E_QKV_DT = 1, E_RESID = 2, E_RELU = 3, E_LM = 4 };

template <int MODE>
__global__ __launch_bounds__(128) void gemm_kernel(const ushort_t* __restrict__ A,
                                                   const ushort_t* __restrict__ Bp,
                                                   int K,
                                                   const float* __restrict__ bias,
                                                   const float* __restrict__ addsrc,
                                                   void* __restrict__ dstv) {
  const int l  = threadIdx.x & 31;
  const int w  = threadIdx.x >> 5;
  const int m0 = (blockIdx.y * 4 + w) * 32;
  const int n0 = blockIdx.x * 64;
  const int KT = K >> 5;
  const size_t PS = (size_t)KT * 512;                // halves per 16-col panel

  const ushort_t* arow0 = A + (size_t)(m0 + (l & 15)) * K + ((l >> 4) * 16);
  const ushort_t* arow1 = arow0 + (size_t)16 * K;
  const ushort_t* bcol  = Bp + (size_t)(n0 >> 4) * PS + (size_t)l * 16;

  v8f acc[2][4] = {{v8f{}, v8f{}, v8f{}, v8f{}}, {v8f{}, v8f{}, v8f{}, v8f{}}};
  for (int kt = 0; kt < KT; ++kt) {
    v16bf a0 = ld_frag(arow0, arow0 + 8);
    v16bf a1 = ld_frag(arow1, arow1 + 8);
    arow0 += 32;
    arow1 += 32;
    const ushort_t* bp = bcol + (size_t)kt * 512;
#pragma unroll
    for (int nc = 0; nc < 4; ++nc) {
      v16bf b = ld_frag(bp, bp + 8);
      acc[0][nc] = wmma_bf16(a0, b, acc[0][nc]);
      acc[1][nc] = wmma_bf16(a1, b, acc[1][nc]);
      bp += PS;
    }
  }

  const int half8 = (l >> 4) * 8, c0 = l & 15;
#pragma unroll
  for (int rb = 0; rb < 2; ++rb) {
#pragma unroll
    for (int nc = 0; nc < 4; ++nc) {
#pragma unroll
      for (int r = 0; r < 8; ++r) {
        int m = m0 + rb * 16 + half8 + r;
        int n = n0 + nc * 16 + c0;
        float v = acc[rb][nc][r];
        if (MODE == E_QKV_TD) {               // -> [B,H,T,D] bf16
          int b_ = m >> 10, t = m & 1023, hh = n >> 6, dd = n & 63;
          ((ushort_t*)dstv)[((size_t)(b_ * Hn + hh) * Tn + t) * Dn + dd] = f2bf(v);
        } else if (MODE == E_QKV_DT) {        // V transposed -> [B,H,D,T] bf16
          int b_ = m >> 10, t = m & 1023, hh = n >> 6, dd = n & 63;
          ((ushort_t*)dstv)[((size_t)(b_ * Hn + hh) * Dn + dd) * Tn + t] = f2bf(v);
        } else if (MODE == E_RESID) {         // fp32: acc + bias + residual
          ((float*)dstv)[(size_t)m * Cn + n] = v + bias[n] + addsrc[(size_t)m * Cn + n];
        } else if (MODE == E_RELU) {          // bf16: relu(acc + bias)
          ((ushort_t*)dstv)[(size_t)m * FFn + n] = f2bf(fmaxf(v + bias[n], 0.f));
        } else {                              // E_LM -> logits fp32 (skip pad cols)
          if (n < Vn) ((float*)dstv)[(size_t)m * Vn + n] = v + bias[n];
        }
      }
    }
  }
}

// ------------------------------------------------- attention scores S = K·Qᵀ
__global__ __launch_bounds__(32) void score_kernel(const ushort_t* __restrict__ kb,
                                                   const ushort_t* __restrict__ qb,
                                                   float* __restrict__ S) {
  if (blockIdx.x > blockIdx.y) return;               // fully-masked tile (wave-uniform)
  const int s0 = blockIdx.x * 16, t0 = blockIdx.y * 16, bh = blockIdx.z;
  const int l  = threadIdx.x;
  const ushort_t* krow = kb + (size_t)(bh * Tn + t0 + (l & 15)) * Dn + ((l >> 4) * 16);
  const ushort_t* qrow = qb + (size_t)(bh * Tn + s0 + (l & 15)) * Dn + ((l >> 4) * 16);
  v16bf a0 = ld_frag(krow, krow + 8);                // d = 0..31
  v16bf a1 = ld_frag(krow + 32, krow + 40);          // d = 32..63
  v16bf b0 = ld_frag(qrow, qrow + 8);
  v16bf b1 = ld_frag(qrow + 32, qrow + 40);
  v8f c = {};
  c = wmma_bf16(a0, b0, c);
  c = wmma_bf16(a1, b1, c);
  const int half8 = (l >> 4) * 8, c0 = l & 15;
#pragma unroll
  for (int r = 0; r < 8; ++r) {
    int t = t0 + half8 + r, s = s0 + c0;
    float v = c[r] * 0.125f;                         // D^-0.5 = 1/8
    if (s > t) v = -1e30f;                           // causal mask
    S[((size_t)(bh * Tn + t)) * Tn + s] = v;
  }
}

// ------------------------------------------------------------- row softmax
__global__ __launch_bounds__(256) void softmax_kernel(const float* __restrict__ S,
                                                      ushort_t* __restrict__ P) {
  const int row = blockIdx.x, tid = threadIdx.x;
  const int t   = row & (Tn - 1);
  const int len = t + 1;
  const float* srow = S + (size_t)row * Tn;
  ushort_t*    prow = P + (size_t)row * Tn;
  __shared__ float red[256];

  float mx = -1e30f;
  for (int s = tid; s < len; s += 256) mx = fmaxf(mx, srow[s]);
  red[tid] = mx; __syncthreads();
  for (int st = 128; st > 0; st >>= 1) {
    if (tid < st) red[tid] = fmaxf(red[tid], red[tid + st]);
    __syncthreads();
  }
  mx = red[0]; __syncthreads();

  float sum = 0.f;
  for (int s = tid; s < len; s += 256) sum += __expf(srow[s] - mx);
  red[tid] = sum; __syncthreads();
  for (int st = 128; st > 0; st >>= 1) {
    if (tid < st) red[tid] += red[tid + st];
    __syncthreads();
  }
  const float inv = 1.f / red[0];

  for (int s = tid; s < Tn; s += 256) {
    float p = (s < len) ? __expf(srow[s] - mx) * inv : 0.f;
    prow[s] = f2bf(p);
  }
}

// ------------------------------------------------------------- O = P · V
// One wave per 32x64 output tile: two 16-row P blocks share the V fragments.
__global__ __launch_bounds__(32) void pv_kernel(const ushort_t* __restrict__ P,
                                                const ushort_t* __restrict__ vtb,
                                                ushort_t* __restrict__ ob) {
  const int t0 = blockIdx.x * 32, bh = blockIdx.y;
  const int l  = threadIdx.x;
  const int b_ = bh / Hn, hh = bh % Hn;
  const ushort_t* prow0 = P + (size_t)(bh * Tn + t0 + (l & 15)) * Tn + ((l >> 4) * 16);
  const ushort_t* prow1 = prow0 + (size_t)16 * Tn;
  const ushort_t* vbase = vtb + (size_t)(bh * Dn) * Tn + ((l >> 4) * 16);
  v8f acc[2][4] = {{v8f{}, v8f{}, v8f{}, v8f{}}, {v8f{}, v8f{}, v8f{}, v8f{}}};
  const int nk = (t0 + 32 + 31) >> 5;                // causal: s < t0+32 (P is 0 beyond)
  for (int ks = 0; ks < nk; ++ks) {
    const int s0 = ks * 32;
    v16bf a0 = ld_frag(prow0 + s0, prow0 + s0 + 8);
    v16bf a1 = ld_frag(prow1 + s0, prow1 + s0 + 8);
#pragma unroll
    for (int nc = 0; nc < 4; ++nc) {
      const ushort_t* vp = vbase + (size_t)(nc * 16 + (l & 15)) * Tn + s0;
      v16bf b = ld_frag(vp, vp + 8);
      acc[0][nc] = wmma_bf16(a0, b, acc[0][nc]);
      acc[1][nc] = wmma_bf16(a1, b, acc[1][nc]);
    }
  }
  const int half8 = (l >> 4) * 8, c0 = l & 15;
#pragma unroll
  for (int rb = 0; rb < 2; ++rb) {
#pragma unroll
    for (int nc = 0; nc < 4; ++nc) {
#pragma unroll
      for (int r = 0; r < 8; ++r) {
        int t = t0 + rb * 16 + half8 + r, d = nc * 16 + c0;  // concat heads -> [B,T,C]
        ob[((size_t)(b_ * Tn + t)) * Cn + hh * Dn + d] = f2bf(acc[rb][nc][r]);
      }
    }
  }
}

// -------------------------------------------------------------- layernorm
__global__ __launch_bounds__(256) void ln_kernel(const float* __restrict__ src,
                                                 const float* __restrict__ g,
                                                 const float* __restrict__ b,
                                                 float* __restrict__ h,
                                                 ushort_t* __restrict__ xb) {
  const int m = blockIdx.x, tid = threadIdx.x;
  const float* row = src + (size_t)m * Cn;
  __shared__ float rs[256], rs2[256];
  float s = 0.f, s2 = 0.f;
  for (int c = tid; c < Cn; c += 256) { float v = row[c]; s += v; s2 += v * v; }
  rs[tid] = s; rs2[tid] = s2; __syncthreads();
  for (int st = 128; st > 0; st >>= 1) {
    if (tid < st) { rs[tid] += rs[tid + st]; rs2[tid] += rs2[tid + st]; }
    __syncthreads();
  }
  const float mean = rs[0] * (1.f / Cn);
  const float var  = rs2[0] * (1.f / Cn) - mean * mean;
  const float inv  = rsqrtf(var + EPSf);
  for (int c = tid; c < Cn; c += 256) {
    float v = (row[c] - mean) * inv * g[c] + b[c];
    h[(size_t)m * Cn + c]  = v;
    xb[(size_t)m * Cn + c] = f2bf(v);
  }
}

// ------------------------------------------------------------------ loss
__global__ __launch_bounds__(256) void rowloss_kernel(const float* __restrict__ logits,
                                                      const int* __restrict__ y,
                                                      float* __restrict__ rowloss) {
  const int m = blockIdx.x, tid = threadIdx.x;
  const float* row = logits + (size_t)m * Vn;
  __shared__ float red[256];
  float mx = -1e30f;
  for (int c = tid; c < Vn; c += 256) mx = fmaxf(mx, row[c]);
  red[tid] = mx; __syncthreads();
  for (int st = 128; st > 0; st >>= 1) {
    if (tid < st) red[tid] = fmaxf(red[tid], red[tid + st]);
    __syncthreads();
  }
  mx = red[0]; __syncthreads();
  float s = 0.f;
  for (int c = tid; c < Vn; c += 256) s += __expf(row[c] - mx);
  red[tid] = s; __syncthreads();
  for (int st = 128; st > 0; st >>= 1) {
    if (tid < st) red[tid] += red[tid + st];
    __syncthreads();
  }
  if (tid == 0) rowloss[m] = (mx + __logf(red[0])) - row[y[m]];
}

__global__ __launch_bounds__(256) void loss_reduce_kernel(const float* __restrict__ rowloss,
                                                          float* __restrict__ out) {
  __shared__ float red[256];
  float s = 0.f;
  for (int i = threadIdx.x; i < Mn; i += 256) s += rowloss[i];
  red[threadIdx.x] = s; __syncthreads();
  for (int st = 128; st > 0; st >>= 1) {
    if (threadIdx.x < st) red[threadIdx.x] += red[threadIdx.x + st];
    __syncthreads();
  }
  if (threadIdx.x == 0) out[0] = red[0] * (1.f / (float)Mn);
}

// =============================================================== launch
extern "C" void kernel_launch(void* const* d_in, const int* in_sizes, int n_in,
                              void* d_out, int out_size, void* d_ws, size_t ws_size,
                              hipStream_t stream) {
  (void)in_sizes; (void)n_in; (void)out_size; (void)ws_size;
  const int*   x     = (const int*)d_in[0];
  const int*   y     = (const int*)d_in[1];
  const float* tok   = (const float*)d_in[2];
  const float* pos   = (const float*)d_in[3];
  const float* Wk    = (const float*)d_in[4];   // [L,H,C,D]
  const float* Wq    = (const float*)d_in[5];
  const float* Wv    = (const float*)d_in[6];
  const float* Wproj = (const float*)d_in[7];   // [L,C,C]
  const float* bproj = (const float*)d_in[8];
  const float* ln1g  = (const float*)d_in[9];
  const float* ln1b  = (const float*)d_in[10];
  const float* W1    = (const float*)d_in[11];  // [L,C,FF]
  const float* b1    = (const float*)d_in[12];
  const float* W2    = (const float*)d_in[13];  // [L,FF,C]
  const float* b2    = (const float*)d_in[14];
  const float* ln2g  = (const float*)d_in[15];
  const float* ln2b  = (const float*)d_in[16];
  const float* Wlm   = (const float*)d_in[17];  // [C,V]
  const float* blm   = (const float*)d_in[18];

  float* logits = (float*)d_out;
  float* lossp  = logits + (size_t)Mn * Vn;

  // ---- workspace carve-up (256B aligned)
  char*  ws  = (char*)d_ws;
  size_t off = 0;
  auto wsalloc = [&](size_t bytes) -> char* {
    char* p = ws + off;
    off = (off + bytes + 255) & ~(size_t)255;
    return p;
  };
  float*    h    = (float*)   wsalloc((size_t)Mn * Cn * 4);
  ushort_t* xb   = (ushort_t*)wsalloc((size_t)Mn * Cn * 2);
  float*    resd = (float*)   wsalloc((size_t)Mn * Cn * 4);
  ushort_t* qb   = (ushort_t*)wsalloc((size_t)Mn * Cn * 2);
  ushort_t* kb   = (ushort_t*)wsalloc((size_t)Mn * Cn * 2);
  ushort_t* vtb  = (ushort_t*)wsalloc((size_t)Mn * Cn * 2);
  float*    S    = (float*)   wsalloc((size_t)Bn * Hn * Tn * Tn * 4);
  ushort_t* P    = (ushort_t*)wsalloc((size_t)Bn * Hn * Tn * Tn * 2);
  ushort_t* ob   = (ushort_t*)wsalloc((size_t)Mn * Cn * 2);
  ushort_t* ffb  = (ushort_t*)wsalloc((size_t)Mn * FFn * 2);
  ushort_t* wqP  = (ushort_t*)wsalloc((size_t)Cn * Cn * 2);
  ushort_t* wkP  = (ushort_t*)wsalloc((size_t)Cn * Cn * 2);
  ushort_t* wvP  = (ushort_t*)wsalloc((size_t)Cn * Cn * 2);
  ushort_t* wpP  = (ushort_t*)wsalloc((size_t)Cn * Cn * 2);
  ushort_t* w1P  = (ushort_t*)wsalloc((size_t)Cn * FFn * 2);
  ushort_t* w2P  = (ushort_t*)wsalloc((size_t)FFn * Cn * 2);
  ushort_t* wlmP = (ushort_t*)wsalloc((size_t)Cn * Vpad * 2);
  float*    rls  = (float*)   wsalloc((size_t)Mn * 4);

  // ---- embedding
  embed_kernel<<<(Mn * Cn) / 256, 256, 0, stream>>>(x, tok, pos, h, xb);

  const dim3 gGemmC(Cn / 64, Mn / 128);    // N = 768
  const dim3 gGemmF(FFn / 64, Mn / 128);   // N = 3072
  const dim3 gScore(Tn / 16, Tn / 16, Bn * Hn);
  const dim3 gPV(Tn / 32, Bn * Hn);

  for (int l = 0; l < Ln; ++l) {
    const size_t wS = (size_t)Cn * Cn;     // per-layer stride for C x C weights
    const size_t fS = (size_t)Cn * FFn;
    // pack weights (bf16, WMMA-B tiled)
    pack_kernel<1><<<(Cn * Cn) / 256, 256, 0, stream>>>(Wq + l * wS, wqP, Cn, Cn);
    pack_kernel<1><<<(Cn * Cn) / 256, 256, 0, stream>>>(Wk + l * wS, wkP, Cn, Cn);
    pack_kernel<1><<<(Cn * Cn) / 256, 256, 0, stream>>>(Wv + l * wS, wvP, Cn, Cn);
    pack_kernel<0><<<(Cn * Cn) / 256, 256, 0, stream>>>(Wproj + l * wS, wpP, Cn, Cn);
    pack_kernel<0><<<(Cn * FFn) / 256, 256, 0, stream>>>(W1 + l * fS, w1P, Cn, FFn);
    pack_kernel<0><<<(FFn * Cn) / 256, 256, 0, stream>>>(W2 + l * fS, w2P, FFn, Cn);

    // q,k,v projections (attention reads raw h: post-norm architecture)
    gemm_kernel<E_QKV_TD><<<gGemmC, 128, 0, stream>>>(xb, wqP, Cn, nullptr, nullptr, qb);
    gemm_kernel<E_QKV_TD><<<gGemmC, 128, 0, stream>>>(xb, wkP, Cn, nullptr, nullptr, kb);
    gemm_kernel<E_QKV_DT><<<gGemmC, 128, 0, stream>>>(xb, wvP, Cn, nullptr, nullptr, vtb);

    // attention: S = K·Qᵀ (reference quirk), softmax, O = P·V
    score_kernel<<<gScore, 32, 0, stream>>>(kb, qb, S);
    softmax_kernel<<<Bn * Hn * Tn, 256, 0, stream>>>(S, P);
    pv_kernel<<<gPV, 32, 0, stream>>>(P, vtb, ob);

    // proj + residual, LN1
    gemm_kernel<E_RESID><<<gGemmC, 128, 0, stream>>>(ob, wpP, Cn, bproj + l * Cn, h, resd);
    ln_kernel<<<Mn, 256, 0, stream>>>(resd, ln1g + l * Cn, ln1b + l * Cn, h, xb);

    // FFN: relu(x W1 + b1) W2 + b2 + residual, LN2
    gemm_kernel<E_RELU><<<gGemmF, 128, 0, stream>>>(xb, w1P, Cn, b1 + l * FFn, nullptr, ffb);
    gemm_kernel<E_RESID><<<gGemmC, 128, 0, stream>>>(ffb, w2P, FFn, b2 + l * Cn, h, resd);
    ln_kernel<<<Mn, 256, 0, stream>>>(resd, ln2g + l * Cn, ln2b + l * Cn, h, xb);
  }

  // LM head -> logits (fp32, directly into d_out), then loss
  pack_kernel<0><<<(Cn * Vpad) / 256, 256, 0, stream>>>(Wlm, wlmP, Cn, Vn);
  gemm_kernel<E_LM><<<dim3(Vpad / 64, Mn / 128), 128, 0, stream>>>(xb, wlmP, Cn, blm,
                                                                   nullptr, logits);
  rowloss_kernel<<<Mn, 256, 0, stream>>>(logits, y, rls);
  loss_reduce_kernel<<<1, 256, 0, stream>>>(rls, lossp);
}
